// TurboQuantAttention_4518305595834
// MI455X (gfx1250) — matrix operand order repaired and checked by reading, compiled, not run on hardware
//
#include <hip/hip_runtime.h>
#include <hip/hip_bf16.h>

// ---------------------------------------------------------------------------
// Problem constants (reference: B=2, S=2048, DIM=2048, H=16, KVH=4, HD=128)
// ---------------------------------------------------------------------------
#define CB    2
#define CS    2048
#define CDIM  2048
#define CH    16
#define CKVH  4
#define CHD   128

typedef __bf16 bf16;
typedef __attribute__((ext_vector_type(16))) __bf16 v16bf;
typedef __attribute__((ext_vector_type(8)))  __bf16 v8bf;
typedef __attribute__((ext_vector_type(8)))  float  v8f;

typedef unsigned int u32x4 __attribute__((ext_vector_type(4)));
typedef int          i32x8 __attribute__((ext_vector_type(8)));
typedef int          i32x4 __attribute__((ext_vector_type(4)));

#if defined(__has_builtin)
#  if __has_builtin(__builtin_amdgcn_tensor_load_to_lds) && \
      __has_builtin(__builtin_amdgcn_s_wait_tensorcnt)
#    define USE_TDM 1
#  else
#    define USE_TDM 0
#  endif
#else
#  define USE_TDM 0
#endif

__device__ __forceinline__ v8f wmma_bf16(v16bf a, v16bf b, v8f c) {
  // D(f32 16x16) = A(bf16 16x32) * B(bf16 32x16) + C
  return __builtin_amdgcn_wmma_f32_16x16x32_bf16(
      /*neg_a=*/false, a, /*neg_b=*/false, b,
      /*c_mod=*/(short)0, c, /*reuse_a=*/false, /*reuse_b=*/false);
}

__device__ __forceinline__ v8f zero8() {
  v8f z;
#pragma unroll
  for (int e = 0; e < 8; ++e) z[e] = 0.0f;
  return z;
}

#if USE_TDM
// Issue one TDM 2D tile load: global (row-major, rowlen elements of bf16 per
// row, row stride = stride_elts) -> LDS at byte offset lds_off.
// D# per CDNA5 ISA ch.8: group0 {count, lds_addr, global_addr, type=2},
// group1 {data_size=2B, tensor_dim0/1, tile_dim0/1, dim0_stride}.
__device__ __forceinline__ void tdm_load_2d(unsigned lds_off,
                                            const void* gptr,
                                            unsigned rowlen_elts,
                                            unsigned nrows,
                                            unsigned stride_elts) {
  unsigned long long ga = (unsigned long long)(uintptr_t)gptr;
  u32x4 g0;
  g0[0] = 1u;                                   // count=1 (valid), user mode
  g0[1] = lds_off;                              // lds_addr (bytes)
  g0[2] = (unsigned)(ga & 0xffffffffu);         // global_addr[31:0]
  g0[3] = (unsigned)((ga >> 32) & 0x01ffffffu)  // global_addr[56:32]
          | (2u << 30);                         // type = 2 ("image")
  i32x8 g1;
  g1[0] = (int)(1u << 16);                      // data_size=1 (2 bytes)
  g1[1] = (int)((rowlen_elts & 0xffffu) << 16); // tensor_dim0[15:0]
  g1[2] = (int)(((rowlen_elts >> 16) & 0xffffu) // tensor_dim0[31:16]
                | ((nrows & 0xffffu) << 16));   // tensor_dim1[15:0]
  g1[3] = (int)(((nrows >> 16) & 0xffffu)       // tensor_dim1[31:16]
                | ((rowlen_elts & 0xffffu) << 16)); // tile_dim0
  g1[4] = (int)(nrows & 0xffffu);               // tile_dim1 (tile_dim2 = 0)
  g1[5] = (int)stride_elts;                     // tensor_dim0_stride[31:0]
  g1[6] = 0;                                    // stride[47:32], dim1_stride lo
  g1[7] = 0;
  i32x4 z;
  z[0] = 0; z[1] = 0; z[2] = 0; z[3] = 0;
#if __clang_major__ >= 23
  i32x8 z8;
#pragma unroll
  for (int e = 0; e < 8; ++e) z8[e] = 0;
  __builtin_amdgcn_tensor_load_to_lds(g0, g1, z, z, z8, 0);
#else
  __builtin_amdgcn_tensor_load_to_lds(g0, g1, z, z, 0);
#endif
}
#endif  // USE_TDM

// A-matrix fragment (16x32 bf16) from row-major LDS tile.
// ISA layout: lanes 0-15 / 16-31 both hold rows 0..15; elements 0..7 at
// k = half*8 + e, elements 8..15 at k = 16 + half*8 + e  (two 16B loads).
__device__ __forceinline__ v16bf lds_fragA(const bf16* base, int stride,
                                           int row16, int koff, int lane) {
  const int half = lane >> 4;
  const bf16* p = base + (row16 + (lane & 15)) * stride + koff + half * 8;
  v8bf lo = *(const v8bf*)(p);
  v8bf hi = *(const v8bf*)(p + 16);
  v16bf r;
#pragma unroll
  for (int e = 0; e < 8; ++e) { r[e] = lo[e]; r[e + 8] = hi[e]; }
  return r;
}

// B-matrix fragment (32x16 bf16). Column n = lane&15, elements 0..7 at
// k = half*8 + e, elements 8..15 at k = 16 + half*8 + e (two 16B loads). LDS tile is [n][k] row-major.
__device__ __forceinline__ v16bf lds_fragB(const bf16* base, int stride,
                                           int n16, int koff, int lane) {
  const int half = lane >> 4;
  const bf16* p = base + (n16 + (lane & 15)) * stride + koff + half * 8;
  v8bf lo = *(const v8bf*)(p);
  v8bf hi = *(const v8bf*)(p + 16);
  v16bf r;
#pragma unroll
  for (int e = 0; e < 8; ++e) { r[e] = lo[e]; r[e + 8] = hi[e]; }
  return r;
}

// ---------------------------------------------------------------------------
// GEMM: C[M,N] = A[M,K] @ W[N,K]^T  (nn.Linear convention)
// Tile 128x128x32, 256 threads = 8 waves (4 row-waves x 2 col-waves),
// each wave computes 2x4 WMMA 16x16 tiles. fp32 operands are converted to
// bf16 while staging into LDS (so no DMA path here); next k-tile is warmed
// into GL2 with global_prefetch.
// HEAD_SPLIT epilogue scatters to [b, head, s, hd] bf16 (fuses transpose).
// ---------------------------------------------------------------------------
template <bool A_BF16, bool OUT_BF16, bool HEAD_SPLIT>
__global__ __launch_bounds__(256) void gemm_xwt(const void* __restrict__ Ap,
                                                const float* __restrict__ W,
                                                void* __restrict__ Cp,
                                                int M, int N, int K,
                                                int S, int HD) {
  __shared__ __align__(16) bf16 As[128 * 32];
  __shared__ __align__(16) bf16 Bs[128 * 32];

  const int tid  = threadIdx.x;
  const int lane = tid & 31;
  const int wid  = tid >> 5;
  const int wrow = wid & 3;   // 4 wave rows * 32 = 128
  const int wcol = wid >> 2;  // 2 wave cols * 64 = 128
  const int m0 = blockIdx.y * 128;
  const int n0 = blockIdx.x * 128;

  v8f acc[2][4];
#pragma unroll
  for (int i = 0; i < 2; ++i)
#pragma unroll
    for (int j = 0; j < 4; ++j) acc[i][j] = zero8();

  for (int k0 = 0; k0 < K; k0 += 32) {
    __syncthreads();
    // ---- stage A tile (convert fp32 -> bf16 on the way in) ----
    if constexpr (!A_BF16) {
      const float* A = (const float*)Ap;
#pragma unroll
      for (int i = 0; i < 4; ++i) {
        int f = i * 256 + tid;          // 1024 float4 chunks
        int row = f >> 3, c4 = f & 7;   // 8 float4 per 32-wide row
        float4 v = *(const float4*)(A + (size_t)(m0 + row) * K + k0 + c4 * 4);
        bf16* d = &As[row * 32 + c4 * 4];
        d[0] = (bf16)v.x; d[1] = (bf16)v.y; d[2] = (bf16)v.z; d[3] = (bf16)v.w;
      }
    } else {
      const bf16* A = (const bf16*)Ap;
#pragma unroll
      for (int i = 0; i < 2; ++i) {
        int f = i * 256 + tid;          // 512 chunks of 8 bf16
        int row = f >> 2, c8 = f & 3;
        *(uint4*)&As[row * 32 + c8 * 8] =
            *(const uint4*)(A + (size_t)(m0 + row) * K + k0 + c8 * 8);
      }
    }
    // ---- stage B tile from W (row n, contraction k) ----
#pragma unroll
    for (int i = 0; i < 4; ++i) {
      int f = i * 256 + tid;
      int row = f >> 3, c4 = f & 7;
      float4 v = *(const float4*)(W + (size_t)(n0 + row) * K + k0 + c4 * 4);
      bf16* d = &Bs[row * 32 + c4 * 4];
      d[0] = (bf16)v.x; d[1] = (bf16)v.y; d[2] = (bf16)v.z; d[3] = (bf16)v.w;
    }
    // ---- warm GL2 with the next k-tile (global_prefetch_b8) ----
    if (k0 + 32 < K) {
      int prow = tid >> 1;  // 2 threads per row cover 128 rows / cachelines
      if constexpr (!A_BF16)
        __builtin_prefetch((const float*)Ap + (size_t)(m0 + prow) * K + k0 + 32,
                           0, 1);
      else
        __builtin_prefetch((const bf16*)Ap + (size_t)(m0 + prow) * K + k0 + 32,
                           0, 1);
      __builtin_prefetch(W + (size_t)(n0 + prow) * K + k0 + 32, 0, 1);
    }
    __syncthreads();

    v16bf afrag[2];
#pragma unroll
    for (int i = 0; i < 2; ++i)
      afrag[i] = lds_fragA(As, 32, wrow * 32 + i * 16, 0, lane);
#pragma unroll
    for (int j = 0; j < 4; ++j) {
      v16bf bfrag = lds_fragB(Bs, 32, wcol * 64 + j * 16, 0, lane);
#pragma unroll
      for (int i = 0; i < 2; ++i)
        acc[i][j] = wmma_bf16(afrag[i], bfrag, acc[i][j]);
    }
  }

  // ---- epilogue ----
  const int half = lane >> 4, l15 = lane & 15;
#pragma unroll
  for (int i = 0; i < 2; ++i)
#pragma unroll
    for (int j = 0; j < 4; ++j)
#pragma unroll
      for (int r = 0; r < 8; ++r) {
        int m = m0 + wrow * 32 + i * 16 + r + half * 8;
        int n = n0 + wcol * 64 + j * 16 + l15;
        float v = acc[i][j][r];
        size_t idx;
        if constexpr (HEAD_SPLIT) {
          int bb = m / S, ss = m % S;
          int hh = n / HD, dd = n % HD;
          int NH = N / HD;
          idx = (((size_t)bb * NH + hh) * S + ss) * (size_t)HD + dd;
        } else {
          idx = (size_t)m * N + n;
        }
        if constexpr (OUT_BF16) ((bf16*)Cp)[idx] = (bf16)v;
        else                    ((float*)Cp)[idx] = v;
      }
}

// ---------------------------------------------------------------------------
// Flash attention (causal, GQA). Grid: (S/128, B*H). 256 threads = 8 waves.
// Each wave owns 16 query rows and the full HD=128 output accumulator
// (8 WMMA C tiles). Q (32KB) and each 64-key K tile (16KB) are contiguous
// bf16 blocks -> staged by the Tensor Data Mover (one descriptor issued by
// wave 0, completion via s_wait_tensorcnt). V needs a transpose in LDS, so
// it keeps the register path.
//
// Static LDS map (byte offsets inside one carved block, 80 KB total):
//   Qs  @     0 : 128x128 bf16 (32 KB)  [qrow][hd]
//   Ks  @ 32768 :  64x128 bf16 (16 KB)  [key][hd]
//   Vt  @ 49152 : 128x64  bf16 (16 KB)  transposed [hd][key]
//   Ps  @ 65536 : 8 x 16x64 bf16 (16 KB) per-wave P tiles
// ---------------------------------------------------------------------------
#define SM_QS 0u
#define SM_KS 32768u
#define SM_VT 49152u
#define SM_PS 65536u

__global__ __launch_bounds__(256) void flash_attn(const bf16* __restrict__ Qp,
                                                  const bf16* __restrict__ Kp,
                                                  const bf16* __restrict__ Vp,
                                                  bf16* __restrict__ O) {
  __shared__ __align__(16) char smem[81920];
  bf16* Qs = (bf16*)(smem + SM_QS);
  bf16* Ks = (bf16*)(smem + SM_KS);
  bf16* Vt = (bf16*)(smem + SM_VT);
  bf16* Pb = (bf16*)(smem + SM_PS);

  const int tid  = threadIdx.x;
  const int lane = tid & 31;
  const int w    = tid >> 5;
  const int half = lane >> 4, l15 = lane & 15;
  bf16* Psw = Pb + w * (16 * 64);

  const int qt = blockIdx.x;
  const int bh = blockIdx.y;         // b*H + h
  const int b  = bh >> 4;            // H = 16
  const int h  = bh & 15;
  const int kvh = h >> 2;            // group size H/KVH = 4

  const size_t q_off  = (size_t)bh * CS * CHD;
  const size_t kv_off = (size_t)(b * CKVH + kvh) * CS * CHD;
  const int q0 = qt * 128;

  // ---- stage Q tile (contiguous 32 KB) ----
#if USE_TDM
  if (w == 0)
    tdm_load_2d(SM_QS, Qp + q_off + (size_t)q0 * CHD, CHD, 128, CHD);
    // wait is folded into the first k-tile's tensorcnt wait (in-order per wave)
#else
#pragma unroll
  for (int i = 0; i < 8; ++i) {
    int f = i * 256 + tid;           // 2048 chunks of 8 bf16
    int row = f >> 4, c = f & 15;
    *(uint4*)&Qs[row * 128 + c * 8] =
        *(const uint4*)(Qp + q_off + (size_t)(q0 + row) * CHD + c * 8);
  }
#endif

  v8f acc[8];
#pragma unroll
  for (int j = 0; j < 8; ++j) acc[j] = zero8();
  float mrow[8], lrow[8];
#pragma unroll
  for (int r = 0; r < 8; ++r) { mrow[r] = -3.0e38f; lrow[r] = 0.0f; }

  const int qmaxw  = q0 + w * 16 + 15;  // last q row this wave owns
  const int ntiles = qt * 2 + 2;        // causal bound: keys <= q0+127
  const float scale = 0.08838834764831845f;  // 1/sqrt(128)

  for (int kt = 0; kt < ntiles; ++kt) {
    const int k0 = kt * 64;
    __syncthreads();  // all readers of Ks/Vt from previous tile are done
    // ---- stage K tile (contiguous 16 KB) via TDM ----
#if USE_TDM
    if (w == 0)
      tdm_load_2d(SM_KS, Kp + kv_off + (size_t)k0 * CHD, CHD, 64, CHD);
#else
#pragma unroll
    for (int i = 0; i < 4; ++i) {
      int f = i * 256 + tid;
      int row = f >> 4, c = f & 15;
      *(uint4*)&Ks[row * 128 + c * 8] =
          *(const uint4*)(Kp + kv_off + (size_t)(k0 + row) * CHD + c * 8);
    }
#endif
    // ---- stage V tile transposed -> [hd][key] (DMA cannot transpose) ----
#pragma unroll
    for (int i = 0; i < 4; ++i) {
      int f = i * 256 + tid;
      int row = f >> 4, c = f & 15;
      uint4 raw = *(const uint4*)(Vp + kv_off + (size_t)(k0 + row) * CHD + c * 8);
      const bf16* e8 = (const bf16*)&raw;
#pragma unroll
      for (int e = 0; e < 8; ++e) Vt[(c * 8 + e) * 64 + row] = e8[e];
    }
#if USE_TDM
    if (w == 0) __builtin_amdgcn_s_wait_tensorcnt(0);
#endif
    __syncthreads();

    if (k0 <= qmaxw) {  // wave-uniform: EXEC stays all-1s for WMMA
      // Q fragments for this wave's 16 rows (4 k-chunks of 32 over HD=128)
      v16bf qf[4];
#pragma unroll
      for (int kc = 0; kc < 4; ++kc)
        qf[kc] = lds_fragA(Qs, 128, w * 16, kc * 32, lane);

      // scores S = Q K^T : 4 key sub-tiles of 16
      v8f sc[4];
#pragma unroll
      for (int n = 0; n < 4; ++n) {
        v8f c = zero8();
#pragma unroll
        for (int kc = 0; kc < 4; ++kc)
          c = wmma_bf16(qf[kc], lds_fragB(Ks, 128, n * 16, kc * 32, lane), c);
        sc[n] = c;
      }

      // scale + causal mask (per-lane select, no divergence)
      const bool need_mask = (k0 + 63) > (q0 + w * 16);
#pragma unroll
      for (int n = 0; n < 4; ++n)
#pragma unroll
        for (int r = 0; r < 8; ++r) {
          float v = sc[n][r] * scale;
          if (need_mask) {
            int qrow = q0 + w * 16 + r + half * 8;
            int krow = k0 + n * 16 + l15;
            if (krow > qrow) v = -3.0e38f;
          }
          sc[n][r] = v;
        }

      // online softmax row stats (width-16 butterflies: lanes 0-15 and 16-31
      // hold disjoint row sets in the WMMA C layout)
      float mnew[8], fac[8];
#pragma unroll
      for (int r = 0; r < 8; ++r) {
        float rm = sc[0][r];
#pragma unroll
        for (int n = 1; n < 4; ++n) rm = fmaxf(rm, sc[n][r]);
        rm = fmaxf(rm, __shfl_xor(rm, 8, 16));
        rm = fmaxf(rm, __shfl_xor(rm, 4, 16));
        rm = fmaxf(rm, __shfl_xor(rm, 2, 16));
        rm = fmaxf(rm, __shfl_xor(rm, 1, 16));
        mnew[r] = fmaxf(mrow[r], rm);
        fac[r]  = __expf(mrow[r] - mnew[r]);
        mrow[r] = mnew[r];
      }

      // P = exp(S - m), row sums, stage P to LDS for A-fragment re-layout
      float rs[8];
#pragma unroll
      for (int r = 0; r < 8; ++r) rs[r] = 0.0f;
#pragma unroll
      for (int n = 0; n < 4; ++n)
#pragma unroll
        for (int r = 0; r < 8; ++r) {
          float sv = sc[n][r];
          float p = (sv > -1.0e37f) ? __expf(sv - mnew[r]) : 0.0f;
          rs[r] += p;
          Psw[(r + half * 8) * 64 + n * 16 + l15] = (bf16)p;
        }
#pragma unroll
      for (int r = 0; r < 8; ++r) {
        float t = rs[r];
        t += __shfl_xor(t, 8, 16);
        t += __shfl_xor(t, 4, 16);
        t += __shfl_xor(t, 2, 16);
        t += __shfl_xor(t, 1, 16);
        lrow[r] = lrow[r] * fac[r] + t;
      }

      // rescale running output
#pragma unroll
      for (int j = 0; j < 8; ++j)
#pragma unroll
        for (int r = 0; r < 8; ++r) acc[j][r] *= fac[r];

      // O += P V  (k-dim = 64 keys = 2 WMMA steps, 8 hd sub-tiles)
#pragma unroll
      for (int ks = 0; ks < 2; ++ks) {
        v16bf pa = lds_fragA(Psw, 64, 0, ks * 32, lane);
#pragma unroll
        for (int j = 0; j < 8; ++j)
          acc[j] = wmma_bf16(pa, lds_fragB(Vt, 64, j * 16, ks * 32, lane),
                             acc[j]);
      }
    }
  }

  // epilogue: normalize, write bf16 [B, S, DIM] (head-merged for O-proj GEMM)
#pragma unroll
  for (int j = 0; j < 8; ++j)
#pragma unroll
    for (int r = 0; r < 8; ++r) {
      int q = q0 + w * 16 + r + half * 8;
      float ov = acc[j][r] / lrow[r];
      O[((size_t)b * CS + q) * CDIM + h * CHD + j * 16 + l15] = (bf16)ov;
    }
}

// ---------------------------------------------------------------------------
// Launch
// ---------------------------------------------------------------------------
extern "C" void kernel_launch(void* const* d_in, const int* in_sizes, int n_in,
                              void* d_out, int out_size, void* d_ws,
                              size_t ws_size, hipStream_t stream) {
  const float* query = (const float*)d_in[0];
  const float* key   = (const float*)d_in[1];
  const float* value = (const float*)d_in[2];
  const float* Wq    = (const float*)d_in[3];
  const float* Wk    = (const float*)d_in[4];
  const float* Wv    = (const float*)d_in[5];
  const float* Wo    = (const float*)d_in[6];

  const int M = CB * CS;  // 4096 token rows

  bf16* Qb = (bf16*)d_ws;                              // [B,H,S,HD]
  bf16* Kb = Qb + (size_t)CB * CH * CS * CHD;          // [B,KVH,S,HD]
  bf16* Vb = Kb + (size_t)CB * CKVH * CS * CHD;
  bf16* Ob = Vb + (size_t)CB * CKVH * CS * CHD;        // [B,S,DIM]

  dim3 blk(256);

  // Q/K/V projections (fp32 A, bf16 head-split output)
  gemm_xwt<false, true, true><<<dim3(CDIM / 128, M / 128), blk, 0, stream>>>(
      query, Wq, Qb, M, CDIM, CDIM, CS, CHD);
  gemm_xwt<false, true, true><<<dim3((CKVH * CHD) / 128, M / 128), blk, 0,
                                stream>>>(key, Wk, Kb, M, CKVH * CHD, CDIM,
                                          CS, CHD);
  gemm_xwt<false, true, true><<<dim3((CKVH * CHD) / 128, M / 128), blk, 0,
                                stream>>>(value, Wv, Vb, M, CKVH * CHD, CDIM,
                                          CS, CHD);

  // causal GQA flash attention (TDM-staged Q/K tiles)
  flash_attn<<<dim3(CS / 128, CB * CH), blk, 0, stream>>>(Qb, Kb, Vb, Ob);

  // output projection (bf16 A, fp32 out)
  gemm_xwt<true, false, false><<<dim3(CDIM / 128, M / 128), blk, 0, stream>>>(
      Ob, Wo, (float*)d_out, M, CDIM, CDIM, CS, CHD);
}
